// NormEMAMSVectorQuantizer_69733089017862
// MI455X (gfx1250) — matrix operand order repaired and checked by reading, not compiled, not run on hardware
//
#include <hip/hip_runtime.h>
#include <hip/hip_bf16.h>

// ---------------- types ----------------
typedef __attribute__((ext_vector_type(16))) _Float16 v16h;
typedef __attribute__((ext_vector_type(8)))  _Float16 v8h;
typedef __attribute__((ext_vector_type(2)))  _Float16 v2h;
typedef __attribute__((ext_vector_type(8)))  float    v8f;

#define N_ROWS   65536      // 16*4096
#define EMBED_D  64
#define N_CODE   8192
#define DECAY    0.99f
#define ONE_M_D  0.01f
#define BETA     0.25f

// workspace byte offsets
#define WS_ZN     0u                         // [N_ROWS][64] f32   16 MiB
#define WS_ZH     16777216u                  // [N_ROWS][64] f16    8 MiB
#define WS_WH     25165824u                  // [N_CODE][64] f16    1 MiB
#define WS_WNORM  26214400u                  // [N_CODE]     f32   32 KiB
#define WS_BINS   26247168u                  // [N_CODE]     f32   32 KiB
#define WS_ESUM   26279936u                  // [N_CODE][64] f32    2 MiB
#define WS_IDX    28377088u                  // [N_ROWS]     i32  256 KiB

// output element offsets (d_out is a float*/int32 region)
#define OUT_ZQ    0u
#define OUT_LOSS  1048576u
#define OUT_IDX   1048577u
#define OUT_CS    1114113u
#define OUT_EA    1122305u
#define OUT_W     1646593u

// ---------------- helpers ----------------
__device__ __forceinline__ float waveReduceSum(float s) {
#pragma unroll
    for (int m = 1; m < 32; m <<= 1) s += __shfl_xor(s, m, 32);
    return s;
}

// ---------------- kernel 1: zero scratch + loss ----------------
__global__ __launch_bounds__(256) void vq_zero_kernel(float* bins, float* esum, float* loss) {
    int i = blockIdx.x * blockDim.x + threadIdx.x;
    int stride = gridDim.x * blockDim.x;
    for (int k = i; k < N_CODE; k += stride) bins[k] = 0.0f;
    for (int k = i; k < N_CODE * EMBED_D; k += stride) esum[k] = 0.0f;
    if (i == 0) *loss = 0.0f;
}

// ---------------- kernel 2: l2norm(z) -> zn (f32), zh (f16) ----------------
__global__ __launch_bounds__(256) void vq_norm_z_kernel(const float* __restrict__ z,
                                                        float* __restrict__ zn,
                                                        _Float16* __restrict__ zh) {
    const int lane = threadIdx.x & 31;
    const int row  = blockIdx.x * 8 + (threadIdx.x >> 5);
    const size_t base = (size_t)row * EMBED_D + lane * 2;
    const float2 v = *(const float2*)(z + base);
    float s = waveReduceSum(v.x * v.x + v.y * v.y);
    const float inv = 1.0f / fmaxf(sqrtf(s), 1e-12f);
    float2 o; o.x = v.x * inv; o.y = v.y * inv;
    *(float2*)(zn + base) = o;
    v2h h; h[0] = (_Float16)o.x; h[1] = (_Float16)o.y;
    *(v2h*)(zh + base) = h;
}

// ---------------- kernel 3: weight -> wh (f16), wnorm = ||w||^2 ----------------
__global__ __launch_bounds__(256) void vq_pack_w_kernel(const float* __restrict__ w,
                                                        _Float16* __restrict__ wh,
                                                        float* __restrict__ wnorm) {
    const int lane = threadIdx.x & 31;
    const int row  = blockIdx.x * 8 + (threadIdx.x >> 5);
    const size_t base = (size_t)row * EMBED_D + lane * 2;
    const float2 v = *(const float2*)(w + base);
    float s = waveReduceSum(v.x * v.x + v.y * v.y);
    if (lane == 0) wnorm[row] = s;
    v2h h; h[0] = (_Float16)v.x; h[1] = (_Float16)v.y;
    *(v2h*)(wh + base) = h;
}

// ---------------- kernel 4: WMMA distance GEMM + argmin ----------------
// one wave per 16-row tile; 32 codes per iteration with ping-pong B-tile sets.
// __builtin_amdgcn_sched_barrier(0) pins the stage order so the register
// allocator cannot coalesce the two sets and the scheduler cannot sink loads
// to just before their WMMA: each load set is covered by a full chunk of
// compute (x4 resident waves) before its s_wait_loadcnt.
__global__ __launch_bounds__(256) void vq_argmin_kernel(const _Float16* __restrict__ zh,
                                                        const _Float16* __restrict__ wh,
                                                        const float* __restrict__ wnorm,
                                                        int* __restrict__ idx_ws,
                                                        int* __restrict__ idx_out) {
    const int lane    = threadIdx.x & 31;
    const int wave    = threadIdx.x >> 5;
    const int rowBase = (blockIdx.x * 8 + wave) * 16;
    const int hl      = lane & 15;    // half-lane
    const int grp     = lane >> 4;    // 0 or 1

    // ---- A tiles: 16x32 f16, contiguous K order (matches B) ----
    // lanes 0-15 : M=hl, V0-7 = K 0..15   (+ tile K-offset)
    // lanes 16-31: M=hl, V0-7 = K 16..31
    const _Float16* zrow = zh + (size_t)(rowBase + hl) * EMBED_D;
    v16h a0, a1;
    {
        v8h lo = *(const v8h*)(zrow + grp * 16);
        v8h hi = *(const v8h*)(zrow + grp * 16 + 8);
#pragma unroll
        for (int i = 0; i < 8; ++i) { a0[i] = lo[i]; a0[i + 8] = hi[i]; }
        lo = *(const v8h*)(zrow + 32 + grp * 16);
        hi = *(const v8h*)(zrow + 32 + grp * 16 + 8);
#pragma unroll
        for (int i = 0; i < 8; ++i) { a1[i] = lo[i]; a1[i + 8] = hi[i]; }
    }

    float bestv[8];
    int   besti[8];
#pragma unroll
    for (int r = 0; r < 8; ++r) { bestv[r] = 3.4e38f; besti[r] = 0; }

    // ---- prologue: set A <- chunk 0 (column = hl, K-half = grp) ----
    v16h b0a, b1a, b0b, b1b;
    float wna, wnb;
    {
        const _Float16* wrow = wh + (size_t)hl * EMBED_D + grp * 16;
        b0a = *(const v16h*)(wrow);
        b1a = *(const v16h*)(wrow + 32);
        wna = wnorm[hl];
    }

    for (int c = 0; c < N_CODE; c += 32) {
        // ================= stage 1: load set B <- chunk c+16 =================
        {
            const _Float16* wrow = wh + (size_t)(c + 16 + hl) * EMBED_D + grp * 16;
            b0b = *(const v16h*)(wrow);
            b1b = *(const v16h*)(wrow + 32);
            wnb = wnorm[c + 16 + hl];
            __builtin_prefetch(wh + (size_t)(((c + 256) & (N_CODE - 1)) + hl) * EMBED_D, 0, 1);
        }
        __builtin_amdgcn_sched_barrier(0);

        // ================= stage 2: compute chunk c from set A ===============
        {
            v8f acc = {};
            acc = __builtin_amdgcn_wmma_f32_16x16x32_f16(false, a0, false, b0a,
                                                         (short)0, acc, false, false);
            acc = __builtin_amdgcn_wmma_f32_16x16x32_f16(false, a1, false, b1a,
                                                         (short)0, acc, false, false);
            const int code = c + hl;   // score = ||w||^2 - 2<z,w>
#pragma unroll
            for (int r = 0; r < 8; ++r) {
                const float score = fmaf(-2.0f, acc[r], wna);
                if (score < bestv[r]) { bestv[r] = score; besti[r] = code; }
            }
        }
        __builtin_amdgcn_sched_barrier(0);

        // ===== stage 3: load set A <- chunk c+32 (wraps harmlessly to 0) =====
        {
            const int cn = (c + 32) & (N_CODE - 1);
            const _Float16* wrow = wh + (size_t)(cn + hl) * EMBED_D + grp * 16;
            b0a = *(const v16h*)(wrow);
            b1a = *(const v16h*)(wrow + 32);
            wna = wnorm[cn + hl];
        }
        __builtin_amdgcn_sched_barrier(0);

        // ================ stage 4: compute chunk c+16 from set B =============
        {
            v8f acc = {};
            acc = __builtin_amdgcn_wmma_f32_16x16x32_f16(false, a0, false, b0b,
                                                         (short)0, acc, false, false);
            acc = __builtin_amdgcn_wmma_f32_16x16x32_f16(false, a1, false, b1b,
                                                         (short)0, acc, false, false);
            const int code = c + 16 + hl;
#pragma unroll
            for (int r = 0; r < 8; ++r) {
                const float score = fmaf(-2.0f, acc[r], wnb);
                if (score < bestv[r]) { bestv[r] = score; besti[r] = code; }
            }
        }
        __builtin_amdgcn_sched_barrier(0);
    }

    // ---- argmin reduce across the 16-lane group (C layout: N = lane%16) ----
#pragma unroll
    for (int m = 1; m < 16; m <<= 1) {
#pragma unroll
        for (int r = 0; r < 8; ++r) {
            const float ov = __shfl_xor(bestv[r], m, 32);
            const int   oi = __shfl_xor(besti[r], m, 32);
            const bool take = (ov < bestv[r]) || (ov == bestv[r] && oi < besti[r]);
            bestv[r] = take ? ov : bestv[r];
            besti[r] = take ? oi : besti[r];
        }
    }
    // C layout rows: grp0 VGPR r -> rowBase+r ; grp1 VGPR r -> rowBase+8+r
#pragma unroll
    for (int r = 0; r < 8; ++r) {
        if (hl == r) {
            const int row = rowBase + grp * 8 + r;
            idx_ws[row]  = besti[r];
            idx_out[row] = besti[r];
        }
    }
}

// ---------------- kernel 5: scatter EMA stats ----------------
__global__ __launch_bounds__(256) void vq_scatter_kernel(const int* __restrict__ idx_ws,
                                                         const float* __restrict__ zn,
                                                         float* __restrict__ bins,
                                                         float* __restrict__ esum) {
    const int lane = threadIdx.x & 31;
    const int row  = blockIdx.x * 8 + (threadIdx.x >> 5);
    const int idx  = idx_ws[row];
    const float2 v = *(const float2*)(zn + (size_t)row * EMBED_D + lane * 2);
    float* dst = esum + (size_t)idx * EMBED_D + lane * 2;
    unsafeAtomicAdd(dst,     v.x);
    unsafeAtomicAdd(dst + 1, v.y);
    if (lane == 0) unsafeAtomicAdd(bins + idx, 1.0f);
}

// ---------------- kernel 6: EMA finalize per code ----------------
__global__ __launch_bounds__(256) void vq_finalize_kernel(const float* __restrict__ bins,
                                                          const float* __restrict__ esum,
                                                          const float* __restrict__ cluster_size,
                                                          const float* __restrict__ embed_avg,
                                                          const float* __restrict__ weight,
                                                          float* __restrict__ out_cs,
                                                          float* __restrict__ out_ea,
                                                          float* __restrict__ out_w) {
    const int lane = threadIdx.x & 31;
    const int row  = blockIdx.x * 8 + (threadIdx.x >> 5);
    const size_t base = (size_t)row * EMBED_D + lane * 2;

    const float b = bins[row];
    if (lane == 0) out_cs[row] = cluster_size[row] * DECAY + ONE_M_D * b;

    const float2 es = *(const float2*)(esum + base);
    const float2 ea = *(const float2*)(embed_avg + base);
    float2 nea; nea.x = ea.x * DECAY + ONE_M_D * es.x;
                nea.y = ea.y * DECAY + ONE_M_D * es.y;
    *(float2*)(out_ea + base) = nea;

    const float bc = (b == 0.0f) ? 1.0f : b;
    float enx = es.x / bc, eny = es.y / bc;
    float s = waveReduceSum(enx * enx + eny * eny);
    const float inv = 1.0f / fmaxf(sqrtf(s), 1e-12f);
    enx *= inv; eny *= inv;

    const float2 w = *(const float2*)(weight + base);
    if (b == 0.0f) { enx = w.x; eny = w.y; }

    float nwx = w.x * DECAY + ONE_M_D * enx;
    float nwy = w.y * DECAY + ONE_M_D * eny;
    s = waveReduceSum(nwx * nwx + nwy * nwy);
    const float inv2 = 1.0f / fmaxf(sqrtf(s), 1e-12f);
    float2 o; o.x = nwx * inv2; o.y = nwy * inv2;
    *(float2*)(out_w + base) = o;
}

// ---------------- kernel 7: z_q gather (STE) + commitment loss ----------------
__global__ __launch_bounds__(256) void vq_zq_loss_kernel(const int* __restrict__ idx_ws,
                                                         const float* __restrict__ zn,
                                                         const float* __restrict__ weight,
                                                         float* __restrict__ out_zq,
                                                         float* __restrict__ loss) {
    const int lane = threadIdx.x & 31;
    const int row  = blockIdx.x * 8 + (threadIdx.x >> 5);
    const int idx  = idx_ws[row];
    const size_t base = (size_t)row * EMBED_D + lane * 2;

    const float2 w = *(const float2*)(weight + (size_t)idx * EMBED_D + lane * 2);
    const float2 z = *(const float2*)(zn + base);
    const float dx = w.x - z.x, dy = w.y - z.y;
    // z_q = z + stopgrad(z_q - z), computed exactly as the reference would round it
    float2 q; q.x = z.x + dx; q.y = z.y + dy;
    *(float2*)(out_zq + base) = q;

    float s = waveReduceSum(dx * dx + dy * dy);
    if (lane == 0)
        unsafeAtomicAdd(loss, s * (BETA / ((float)N_ROWS * (float)EMBED_D)));
}

// ---------------- launch ----------------
extern "C" void kernel_launch(void* const* d_in, const int* in_sizes, int n_in,
                              void* d_out, int out_size, void* d_ws, size_t ws_size,
                              hipStream_t stream) {
    const float* z            = (const float*)d_in[0];
    const float* weight       = (const float*)d_in[1];
    const float* cluster_size = (const float*)d_in[2];
    const float* embed_avg    = (const float*)d_in[3];

    char* ws = (char*)d_ws;
    float*    zn    = (float*)(ws + WS_ZN);
    _Float16* zh    = (_Float16*)(ws + WS_ZH);
    _Float16* wh    = (_Float16*)(ws + WS_WH);
    float*    wnorm = (float*)(ws + WS_WNORM);
    float*    bins  = (float*)(ws + WS_BINS);
    float*    esum  = (float*)(ws + WS_ESUM);
    int*      idxw  = (int*)(ws + WS_IDX);

    float* outF    = (float*)d_out;
    float* out_zq  = outF + OUT_ZQ;
    float* out_ls  = outF + OUT_LOSS;
    int*   out_idx = (int*)(outF + OUT_IDX);
    float* out_cs  = outF + OUT_CS;
    float* out_ea  = outF + OUT_EA;
    float* out_w   = outF + OUT_W;

    vq_zero_kernel<<<1024, 256, 0, stream>>>(bins, esum, out_ls);
    vq_norm_z_kernel<<<N_ROWS / 8, 256, 0, stream>>>(z, zn, zh);
    vq_pack_w_kernel<<<N_CODE / 8, 256, 0, stream>>>(weight, wh, wnorm);
    vq_argmin_kernel<<<N_ROWS / 16 / 8, 256, 0, stream>>>(zh, wh, wnorm, idxw, out_idx);
    vq_scatter_kernel<<<N_ROWS / 8, 256, 0, stream>>>(idxw, zn, bins, esum);
    vq_finalize_kernel<<<N_CODE / 8, 256, 0, stream>>>(bins, esum, cluster_size,
                                                       embed_avg, weight,
                                                       out_cs, out_ea, out_w);
    vq_zq_loss_kernel<<<N_ROWS / 8, 256, 0, stream>>>(idxw, zn, weight, out_zq, out_ls);
}
